// GRUEncoder_66331474919459
// MI455X (gfx1250) — matrix operand order repaired and checked by reading, not compiled, not run on hardware
//
#include <hip/hip_runtime.h>

#define TT 12
#define NN 325
#define FF 64
#define K2 128          // 2F
#define NF (NN * FF)    // 20800
#define TNF (TT * NF)   // 249600

typedef __attribute__((ext_vector_type(16))) __bf16       v16bf;
typedef __attribute__((ext_vector_type(4)))  __bf16       v4bf;
typedef __attribute__((ext_vector_type(8)))  float        v8f;
typedef __attribute__((ext_vector_type(4)))  float        v4f;
typedef __attribute__((ext_vector_type(4)))  unsigned int v4u;

union Frag {
    v16bf v;
    v4bf  b4[4];
    v4u   q[2];
};

__device__ __forceinline__ float fast_sigmoid(float x) {
    float e = __builtin_amdgcn_exp2f(-1.44269504088896340736f * x);
    return __builtin_amdgcn_rcpf(1.0f + e);
}

__device__ __forceinline__ float fast_tanh(float x) {
    float e = __builtin_amdgcn_exp2f(-2.88539008177792681472f * x);
    return 2.0f * __builtin_amdgcn_rcpf(1.0f + e) - 1.0f;
}

// Pre-pass: convert all per-step weights to bf16 once into workspace.
__global__ void wconv_kernel(const float* __restrict__ fw,
                             const float* __restrict__ zw,
                             __bf16* __restrict__ ws) {
    int i = blockIdx.x * blockDim.x + threadIdx.x;
    const int nfw = TT * K2 * K2;   // 196608
    const int nzw = TT * FF * K2;   // 98304
    if (i < nfw)            ws[i] = (__bf16)fw[i];
    else if (i < nfw + nzw) ws[i] = (__bf16)zw[i - nfw];
}

__launch_bounds__(256)
__global__ void gru_kernel(const float* __restrict__ x,
                           const float* __restrict__ ff_b,
                           const float* __restrict__ zff_b,
                           const __bf16* __restrict__ fwbf,
                           const __bf16* __restrict__ zwbf,
                           float* __restrict__ out) {
    __shared__ __align__(16) __bf16 s_fw[K2 * K2];     // 32 KB
    __shared__ __align__(16) __bf16 s_zw[FF * K2];     // 16 KB
    __shared__ __align__(16) __bf16 s_h[8 * 16 * 64];  // 16 KB (h, then r*h)

    const int tid  = threadIdx.x;
    const int wave = tid >> 5;
    const int lane = tid & 31;
    const int lh   = lane & 15;
    const int hi   = lane >> 4;

    const int rwb = blockIdx.x * 128 + wave * 16;   // wave's 16-row tile base

    // A-layout: this lane supplies row rwb+lh of the A matrix.
    int rA = rwb + lh;
    int bA = rA / NN;
    int nA = rA - bA * NN;
    const int xbase = bA * TNF + nA * FF;

    // D-layout: this lane holds rows rwb + hi*8 + j (j = VGPR index 0..7).
    int obase[8];
    #pragma unroll
    for (int j = 0; j < 8; ++j) {
        int rD = rwb + hi * 8 + j;
        int bD = rD / NN;
        int nD = rD - bD * NN;
        obase[j] = bD * TNF + nD * FF;
    }

    v8f hD[4];   // h state, f32, D-layout: hD[c][j] = h[row hi*8+j][col c*16+lh]
    #pragma unroll
    for (int c = 0; c < 4; ++c)
        #pragma unroll
        for (int j = 0; j < 8; ++j) hD[c][j] = 0.0f;

    __bf16* s_hw = s_h + wave * 1024;

    #pragma unroll 1
    for (int t = 0; t < TT; ++t) {
        // ---- stage this step's bf16 weights into LDS ----
        __syncthreads();
        {
            const v4u* src = (const v4u*)(fwbf + t * (K2 * K2));
            v4u* dst = (v4u*)s_fw;
            #pragma unroll
            for (int i = 0; i < 8; ++i) dst[tid + i * 256] = src[tid + i * 256];
            const v4u* src2 = (const v4u*)(zwbf + t * (FF * K2));
            v4u* dst2 = (v4u*)s_zw;
            #pragma unroll
            for (int i = 0; i < 4; ++i) dst2[tid + i * 256] = src2[tid + i * 256];
        }
        __syncthreads();

        // ---- write h (bf16) into per-wave LDS tile (D-layout -> row-major) ----
        #pragma unroll
        for (int c = 0; c < 4; ++c)
            #pragma unroll
            for (int j = 0; j < 8; ++j)
                s_hw[(hi * 8 + j) * 64 + c * 16 + lh] = (__bf16)hD[c][j];

        // ---- A-fragments: k 0..63 = x (global f32->bf16), k 64..127 = h (LDS) ----
        const float* xp = x + xbase + t * NF;
        Frag a[4];
        #pragma unroll
        for (int kt = 0; kt < 2; ++kt) {
            int c0 = kt * 32 + hi * 8;
            v4f p0 = __builtin_nontemporal_load((const v4f*)(xp + c0));
            v4f p1 = __builtin_nontemporal_load((const v4f*)(xp + c0 + 4));
            v4f p2 = __builtin_nontemporal_load((const v4f*)(xp + c0 + 16));
            v4f p3 = __builtin_nontemporal_load((const v4f*)(xp + c0 + 20));
            a[kt].b4[0] = __builtin_convertvector(p0, v4bf);
            a[kt].b4[1] = __builtin_convertvector(p1, v4bf);
            a[kt].b4[2] = __builtin_convertvector(p2, v4bf);
            a[kt].b4[3] = __builtin_convertvector(p3, v4bf);
        }
        #pragma unroll
        for (int kt = 0; kt < 2; ++kt) {
            int idx = lh * 64 + kt * 32 + hi * 8;
            a[2 + kt].q[0] = *(const v4u*)(s_hw + idx);
            a[2 + kt].q[1] = *(const v4u*)(s_hw + idx + 16);
        }

        // ---- GEMM1: ru = sigmoid([x|h] @ fw^T + fb)  (16 x 128 tile) ----
        v8f ru[8];
        #pragma unroll
        for (int o = 0; o < 8; ++o) {
            float fb = ff_b[t * K2 + o * 16 + lh];
            v8f acc;
            #pragma unroll
            for (int j = 0; j < 8; ++j) acc[j] = fb;
            #pragma unroll
            for (int kt = 0; kt < 4; ++kt) {
                Frag bfr;
                int idx = (o * 16 + lh) * K2 + kt * 32 + hi * 8;
                bfr.q[0] = *(const v4u*)(s_fw + idx);
                bfr.q[1] = *(const v4u*)(s_fw + idx + 16);
                acc = __builtin_amdgcn_wmma_f32_16x16x32_bf16(
                    false, a[kt].v, false, bfr.v, (short)0, acc, false, false);
            }
            #pragma unroll
            for (int j = 0; j < 8; ++j) acc[j] = fast_sigmoid(acc[j]);
            ru[o] = acc;
        }

        // ---- r*h (bf16) into the same per-wave LDS tile ----
        #pragma unroll
        for (int c = 0; c < 4; ++c)
            #pragma unroll
            for (int j = 0; j < 8; ++j)
                s_hw[(hi * 8 + j) * 64 + c * 16 + lh] = (__bf16)(ru[c][j] * hD[c][j]);

        #pragma unroll
        for (int kt = 0; kt < 2; ++kt) {
            int idx = lh * 64 + kt * 32 + hi * 8;
            a[2 + kt].q[0] = *(const v4u*)(s_hw + idx);
            a[2 + kt].q[1] = *(const v4u*)(s_hw + idx + 16);
        }

        // ---- GEMM2: z = tanh([x|r*h] @ zw^T + zb); h' = u*z + (1-u)*h ----
        #pragma unroll
        for (int o = 0; o < 4; ++o) {
            float zb = zff_b[t * FF + o * 16 + lh];
            v8f acc;
            #pragma unroll
            for (int j = 0; j < 8; ++j) acc[j] = zb;
            #pragma unroll
            for (int kt = 0; kt < 4; ++kt) {
                Frag bfr;
                int idx = (o * 16 + lh) * K2 + kt * 32 + hi * 8;
                bfr.q[0] = *(const v4u*)(s_zw + idx);
                bfr.q[1] = *(const v4u*)(s_zw + idx + 16);
                acc = __builtin_amdgcn_wmma_f32_16x16x32_bf16(
                    false, a[kt].v, false, bfr.v, (short)0, acc, false, false);
            }
            #pragma unroll
            for (int j = 0; j < 8; ++j) {
                float z  = fast_tanh(acc[j]);
                float u  = ru[4 + o][j];
                float hn = hD[o][j] + u * (z - hD[o][j]);
                hD[o][j] = hn;
                __builtin_nontemporal_store(hn, &out[obase[j] + t * NF + o * 16 + lh]);
            }
        }
    }
}

extern "C" void kernel_launch(void* const* d_in, const int* in_sizes, int n_in,
                              void* d_out, int out_size, void* d_ws, size_t ws_size,
                              hipStream_t stream) {
    (void)in_sizes; (void)n_in; (void)out_size; (void)ws_size;
    const float* x     = (const float*)d_in[0];
    const float* ff_w  = (const float*)d_in[1];
    const float* ff_b  = (const float*)d_in[2];
    const float* zff_w = (const float*)d_in[3];
    const float* zff_b = (const float*)d_in[4];
    float* out = (float*)d_out;

    __bf16* wsbf = (__bf16*)d_ws;               // needs 589,824 bytes
    const int nconv = TT * (K2 * K2 + FF * K2); // 294912
    wconv_kernel<<<(nconv + 255) / 256, 256, 0, stream>>>(ff_w, zff_w, wsbf);
    gru_kernel<<<325, 256, 0, stream>>>(x, ff_b, zff_b,
                                        wsbf, wsbf + TT * K2 * K2, out);
}
